// WordleEmbeddingModel_27539330302402
// MI455X (gfx1250) — matrix operand order repaired and checked by reading, not compiled, not run on hardware
//
#include <hip/hip_runtime.h>
#include <hip/hip_bf16.h>

typedef __attribute__((ext_vector_type(16))) __bf16 v16bf;
typedef __attribute__((ext_vector_type(8)))  __bf16 v8bf;
typedef __attribute__((ext_vector_type(8)))  float  v8f;

#define ROWS 128   // batch rows per block
#define HID  256
#define FEAT 96
#define RSTR 104   // padded row stride (halves) for LDS combined matrix

// ---------------------------------------------------------------------------
// Kernel 0: convert W1 [96,256] f32 -> bf16, pre-swizzled into WMMA B-operand
// fragments. Fragment (t,s) covers K rows s*32..s*32+31, N cols t*16..t*16+15.
// B order used here (contiguous k, same as A): lane l holds col n=l&15, K = i + 16*(l>>4) for
// half-index i=0..15. Stored as 48 fragments x 32 lanes x 16 halves (48 KB).
// ---------------------------------------------------------------------------
__global__ __launch_bounds__(256) void prep_w1_bf16(const float* __restrict__ W1,
                                                    __bf16* __restrict__ ws) {
    for (int idx = threadIdx.x; idx < 48 * 512; idx += 256) {
        int frag = idx >> 9;          // 0..47
        int r    = idx & 511;
        int lane = r >> 4;            // 0..31
        int i    = r & 15;            // half index within lane
        int t    = frag / 3;          // column tile
        int s    = frag % 3;          // K step
        int hi   = lane >> 4;
        int nn   = lane & 15;
        int k    = s * 32 + i + 16 * hi;   // 0..95
        int n    = t * 16 + nn;            // 0..255
        ws[idx] = (__bf16)W1[k * HID + n];
    }
}

// ---------------------------------------------------------------------------
// Main kernel: fused embedding-gather + masked means + (x@W1+b1) ReLU @W2+b2.
// 256 threads = 8 waves; each wave owns a 16-row tile; 48 bf16 WMMAs/wave.
// ---------------------------------------------------------------------------
__global__ __launch_bounds__(256) void wordle_mlp_kernel(
    const int* __restrict__ gidx, const int* __restrict__ cidx,
    const int* __restrict__ pidx, const int* __restrict__ plen,
    const int* __restrict__ aidx, const int* __restrict__ alen,
    const float* __restrict__ gtab, const float* __restrict__ ctab,
    const __bf16* __restrict__ w1f, const float* __restrict__ b1,
    const float* __restrict__ W2, const float* __restrict__ b2,
    float* __restrict__ out, int Btot)
{
    __shared__ __align__(16) __bf16 comb[ROWS * RSTR];   // 26,624 B
    __shared__ __align__(16) float  red[8 * 32 * 8];     //  8,192 B

    const int  tid     = threadIdx.x;
    const long rowbase = (long)blockIdx.x * ROWS;

    // ---- Phase 1: build combined feature matrix [ROWS x 96] in LDS (bf16) --
    for (int e = tid; e < ROWS * FEAT; e += 256) {
        int  r    = e / FEAT;
        int  f    = e - r * FEAT;
        long grow = rowbase + r;
        float val = 0.0f;
        if (grow < Btot) {
            if (f < 80) {                       // guess_flat / constraint_flat
                int f2 = (f < 40) ? f : (f - 40);
                int j  = f2 >> 3;
                int d  = f2 & 7;
                if (f < 40) val = gtab[gidx[grow * 5 + j] * 8 + d];
                else        val = ctab[cidx[grow * 5 + j] * 8 + d];
            } else {                            // masked means (presence/absent)
                int d = f & 7;
                const int* ip = (f < 88) ? (pidx + grow * 10) : (aidx + grow * 10);
                int ln = (f < 88) ? plen[grow] : alen[grow];
                float s = 0.0f;
#pragma unroll
                for (int k = 0; k < 10; ++k) {
                    if (k < ln) s += gtab[ip[k] * 8 + d];
                }
                val = (ln > 0) ? (s / (float)ln) : 0.0f;
            }
        }
        comb[r * RSTR + f] = (__bf16)val;
    }
    __syncthreads();

    // ---- Phase 2: WMMA layer 1 with fused bias/ReLU/W2 epilogue ------------
    const int w    = tid >> 5;       // wave 0..7
    const int l    = tid & 31;       // lane
    const int m    = l & 15;         // row within tile / N column
    const int hi   = l >> 4;
    const int rloc = w * 16 + m;

    // A fragments: contiguous k order used here (same as B): halves 0..7 -> K = k0+16*hi..,
    //              halves 8..15 -> K = k0+16*hi+8..  (contiguous b128 chunks)
    v16bf A[3];
#pragma unroll
    for (int s = 0; s < 3; ++s) {
        const __bf16* p = &comb[rloc * RSTR + s * 32 + 16 * hi];
        v8bf lo  = *(const v8bf*)p;
        v8bf hi8 = *(const v8bf*)(p + 8);
#pragma unroll
        for (int i = 0; i < 8; ++i) { A[s][i] = lo[i]; A[s][i + 8] = hi8[i]; }
    }

    v8f acc = {};
#pragma unroll
    for (int t = 0; t < 16; ++t) {
        v8f C = {};
#pragma unroll
        for (int s = 0; s < 3; ++s) {
            const __bf16* bp = w1f + (t * 3 + s) * 512 + l * 16;  // pre-swizzled
            v8bf blo = *(const v8bf*)bp;
            v8bf bhi = *(const v8bf*)(bp + 8);
            v16bf Bf;
#pragma unroll
            for (int i = 0; i < 8; ++i) { Bf[i] = blo[i]; Bf[i + 8] = bhi[i]; }
            C = __builtin_amdgcn_wmma_f32_16x16x32_bf16(
                    false, A[s], false, Bf, (short)0, C, false, false);
        }
        // C layout: lane l holds (rows r + 8*hi, col n = t*16 + m), r = VGPR idx
        int   n   = t * 16 + m;
        float b1v = b1[n];
        float w2v = W2[n];
#pragma unroll
        for (int r = 0; r < 8; ++r) {
            float h = C[r] + b1v;
            h = (h > 0.0f) ? h : 0.0f;
            acc[r] += h * w2v;
        }
    }

    // ---- Phase 3: reduce 16 lanes per hi-group, add b2, store --------------
    *(v8f*)&red[(w * 32 + l) * 8] = acc;
    __syncthreads();

    if (l < 16) {
        int rr = l & 7;          // VGPR index
        int hh = l >> 3;         // hi group: out row = rr + 8*hh = l
        float sum = b2[0];
#pragma unroll
        for (int q = 0; q < 16; ++q) sum += red[(w * 32 + hh * 16 + q) * 8 + rr];
        long orow = rowbase + w * 16 + l;
        if (orow < Btot) out[orow] = sum;
    }
}

extern "C" void kernel_launch(void* const* d_in, const int* in_sizes, int n_in,
                              void* d_out, int out_size, void* d_ws, size_t ws_size,
                              hipStream_t stream) {
    const int*   gidx = (const int*)d_in[0];
    const int*   cidx = (const int*)d_in[1];
    const int*   pidx = (const int*)d_in[2];
    const int*   plen = (const int*)d_in[3];
    const int*   aidx = (const int*)d_in[4];
    const int*   alen = (const int*)d_in[5];
    const float* gtab = (const float*)d_in[6];
    const float* ctab = (const float*)d_in[7];
    const float* W1   = (const float*)d_in[8];
    const float* b1   = (const float*)d_in[9];
    const float* W2   = (const float*)d_in[10];
    const float* b2   = (const float*)d_in[11];
    float*       out  = (float*)d_out;

    const int Btot = in_sizes[0] / 5;
    __bf16* ws = (__bf16*)d_ws;   // needs 48 KB for swizzled bf16 W1

    prep_w1_bf16<<<1, 256, 0, stream>>>(W1, ws);

    const int blocks = (Btot + ROWS - 1) / ROWS;
    wordle_mlp_kernel<<<blocks, 256, 0, stream>>>(
        gidx, cidx, pidx, plen, aidx, alen, gtab, ctab,
        ws, b1, W2, b2, out, Btot);
}
